// FasterDiTStage_50972671869512
// MI455X (gfx1250) — matrix operand order repaired and checked by reading, not compiled, not run on hardware
//
#include <hip/hip_runtime.h>
#include <math.h>

// ---------- types ----------
typedef _Float16 f16;
typedef __attribute__((ext_vector_type(16))) _Float16 v16h;
typedef __attribute__((ext_vector_type(8)))  float    v8f;
typedef __attribute__((ext_vector_type(4)))  unsigned int u32x4;

union Frag { v16h h; u32x4 u[2]; };

#define HEADS  8
#define CDIM   512
#define HD     64
#define NB     8
#define BCT    512          // total carrier tokens (8 batches * 64)
#define NWIN   512          // number of windows
#define NTOK   65           // tokens per window (1 ct + 64 px)
#define MWIN   (NWIN*NTOK)  // 33280 rows, = 520 * 64
#define HIDDEN 2048

__device__ __forceinline__ float gelu_exact(float x) {
    return 0.5f * x * (1.0f + erff(x * 0.70710678118654752f));
}

__device__ __forceinline__ v8f wmma_f16(const Frag& a, const Frag& b, v8f c) {
    return __builtin_amdgcn_wmma_f32_16x16x32_f16(false, a.h, false, b.h,
                                                  (short)0, c, false, false);
}

// CDNA5 async memory->LDS copy (ASYNCcnt tracked). VDST VGPR carries the LDS
// byte offset (= low 32 bits of the flat shared address per the aperture map).
__device__ __forceinline__ void async_load_b128(const void* g, void* lds) {
    unsigned l = (unsigned)(unsigned long long)lds;
    asm volatile("global_load_async_to_lds_b128 %0, %1, off"
                 :: "v"(l), "v"(g)
                 : "memory");
}
__device__ __forceinline__ void wait_async0() {
    asm volatile("s_wait_asynccnt 0x0" ::: "memory");
}

// ---------- generic WMMA GEMM:  C[M,N] = A[M,K](f16) * Wt[N,K](f16)^T ----------
// 128 threads (4 waves), 64x64 block tile, K-step 32, double-buffered LDS fed by
// global_load_async_to_lds_b128; one barrier per k-step.
enum { GF_BIAS = 1, GF_GELU = 2, GF_OUTF16 = 4, GF_RESID = 8 };

template<int FLAGS>
__global__ __launch_bounds__(128)
void gemm_f16_kernel(const f16* __restrict__ A, const f16* __restrict__ Wt,
                     const float* __restrict__ bias, f16* __restrict__ out16,
                     float* __restrict__ resid, int M, int N, int K)
{
    __shared__ __align__(16) f16 sA[2][64 * 32];
    __shared__ __align__(16) f16 sB[2][64 * 32];
    const int tid  = threadIdx.x;
    const int wv   = tid >> 5;        // wave 0..3 -> row strip
    const int lane = tid & 31;
    const int nl   = lane & 15;
    const int hi2  = lane >> 4;       // lane half
    const long bm = (long)blockIdx.x * 64;
    const long bn = (long)blockIdx.y * 64;

    // each thread owns two 8-half chunks of the 64x32 tile
    const int ch0  = tid,        ch1  = tid + 128;
    const int row0 = ch0 >> 2,   off0 = (ch0 & 3) * 8;
    const int row1 = ch1 >> 2,   off1 = (ch1 & 3) * 8;

    auto issue = [&](int buf, int k0) {
        async_load_b128(&A [(bm + row0) * K + k0 + off0], &sA[buf][row0 * 32 + off0]);
        async_load_b128(&A [(bm + row1) * K + k0 + off1], &sA[buf][row1 * 32 + off1]);
        async_load_b128(&Wt[(bn + row0) * K + k0 + off0], &sB[buf][row0 * 32 + off0]);
        async_load_b128(&Wt[(bn + row1) * K + k0 + off1], &sB[buf][row1 * 32 + off1]);
    };

    v8f acc[4] = {};
    issue(0, 0);
    int buf = 0;
    for (int k0 = 0; k0 < K; k0 += 32) {
        wait_async0();          // own async stores into LDS done
        __syncthreads();        // everyone's tile visible; prev tile fully consumed
        if (k0 + 32 < K) issue(buf ^ 1, k0 + 32);   // prefetch next tile async

        Frag af;
        af.u[0] = *(const u32x4*)&sA[buf][(16 * wv + nl) * 32 + hi2 * 8];
        af.u[1] = *(const u32x4*)&sA[buf][(16 * wv + nl) * 32 + hi2 * 8 + 16];
        #pragma unroll
        for (int j = 0; j < 4; j++) {
            Frag bf;
            bf.u[0] = *(const u32x4*)&sB[buf][(j * 16 + nl) * 32 + hi2 * 8];
            bf.u[1] = *(const u32x4*)&sB[buf][(j * 16 + nl) * 32 + hi2 * 8 + 16];
            acc[j] = wmma_f16(af, bf, acc[j]);
        }
        buf ^= 1;
    }

    #pragma unroll
    for (int j = 0; j < 4; j++) {
        int n = (int)bn + j * 16 + nl;
        #pragma unroll
        for (int r = 0; r < 8; r++) {
            long m = bm + 16 * wv + r + hi2 * 8;
            float v = acc[j][r];
            if (FLAGS & GF_BIAS)   v += bias[n];
            if (FLAGS & GF_GELU)   v = gelu_exact(v);
            if (FLAGS & GF_OUTF16) out16[m * N + n] = (f16)v;
            if (FLAGS & GF_RESID)  resid[m * N + n] += v;
        }
    }
}

// ---------- layernorm: f32 row (512) -> f16 row ----------
__global__ __launch_bounds__(128)
void ln_kernel(const float* __restrict__ x, const float* __restrict__ g,
               const float* __restrict__ b, f16* __restrict__ y)
{
    const long row = blockIdx.x;
    const float* xr = x + row * CDIM;
    const int tid = threadIdx.x;
    float s = 0.f, s2 = 0.f;
    #pragma unroll
    for (int i = 0; i < 4; i++) {
        float v = xr[tid + 128 * i];
        s += v; s2 += v * v;
    }
    __shared__ float r1[128], r2[128];
    r1[tid] = s; r2[tid] = s2;
    __syncthreads();
    for (int off = 64; off; off >>= 1) {
        if (tid < off) { r1[tid] += r1[tid + off]; r2[tid] += r2[tid + off]; }
        __syncthreads();
    }
    float mean = r1[0] * (1.f / 512.f);
    float var  = r2[0] * (1.f / 512.f) - mean * mean;
    float inv  = rsqrtf(var + 1e-5f);
    #pragma unroll
    for (int i = 0; i < 4; i++) {
        int c = tid + 128 * i;
        y[row * CDIM + c] = (f16)((xr[c] - mean) * inv * g[c] + b[c]);
    }
}

// ---------- weight convert+transpose: src[K][N] f32 -> dst[N][K] f16 ----------
__global__ void wcvt_kernel(const float* __restrict__ src, f16* __restrict__ dst,
                            int K, int N)
{
    long i = (long)blockIdx.x * 256 + threadIdx.x;
    if (i >= (long)K * N) return;
    long n = i / K, k = i % K;
    dst[i] = (f16)src[k * (long)N + n];
}

// ---------- RoPE tables (64 positions x 32 pairs) ----------
__global__ void rope_tab_kernel(float* __restrict__ cosT, float* __restrict__ sinT)
{
    int i = blockIdx.x * 256 + threadIdx.x;
    if (i >= 64 * 32) return;
    int p = i >> 5, j = i & 31;
    int jj = j & 15;
    float freq = powf(10000.f, -(float)jj / 16.f);
    float pos  = (j < 16) ? (float)(p >> 3) : (float)(p & 7);
    float ang  = pos * freq;
    cosT[i] = cosf(ang);
    sinT[i] = sinf(ang);
}

// ---------- RoPE apply in-place on ct q,k (f16 qkv rows 0..511) ----------
__global__ void rope_apply_kernel(f16* __restrict__ qkv,
                                  const float* __restrict__ cosT,
                                  const float* __restrict__ sinT)
{
    int i = blockIdx.x * 256 + threadIdx.x;
    if (i >= BCT * HEADS * 32 * 2) return;
    int pair = i & 31; int t = i >> 5;
    int h  = t & 7;  t >>= 3;
    int qk = t & 1;  t >>= 1;
    int row = t;                               // 0..511
    f16* p = qkv + (long)row * 1536 + qk * 512 + h * 64 + pair * 2;
    float c = cosT[(row & 63) * 32 + pair];
    float s = sinT[(row & 63) * 32 + pair];
    float a = (float)p[0], bb = (float)p[1];
    p[0] = (f16)(a * c - bb * s);
    p[1] = (f16)(a * s + bb * c);
}

// ---------- carrier-token cosine attention (per (batch, head)) ----------
__global__ __launch_bounds__(128)
void ct_attn_kernel(const f16* __restrict__ qkv, const float* __restrict__ ls,
                    float* __restrict__ ct)
{
    __shared__ __align__(16) f16 sQ [64 * 64];   // qn * scale, reused as P
    __shared__ __align__(16) f16 sK [64 * 64];   // kn
    __shared__ __align__(16) f16 sVt[64 * 64];   // V^T  [d][t]
    const int b = blockIdx.x >> 3, h = blockIdx.x & 7;
    const int tid = threadIdx.x, wv = tid >> 5, lane = tid & 31;
    const int nl = lane & 15, hi2 = lane >> 4;
    const float scale = __expf(fminf(ls[h], 4.6052f));
    const f16* base = qkv + (long)b * 64 * 1536;

    if (tid < 64) {
        const f16* qr = base + (long)tid * 1536 + h * 64;
        float ss = 0.f;
        for (int d = 0; d < 64; d++) { float v = (float)qr[d]; ss += v * v; }
        float inv = scale / fmaxf(sqrtf(ss), 1e-12f);
        for (int d = 0; d < 64; d++) sQ[tid * 64 + d] = (f16)((float)qr[d] * inv);
    } else {
        int t = tid - 64;
        const f16* kr = base + (long)t * 1536 + 512 + h * 64;
        float ss = 0.f;
        for (int d = 0; d < 64; d++) { float v = (float)kr[d]; ss += v * v; }
        float inv = 1.f / fmaxf(sqrtf(ss), 1e-12f);
        for (int d = 0; d < 64; d++) sK[t * 64 + d] = (f16)((float)kr[d] * inv);
    }
    for (int i = tid; i < 64 * 64; i += 128) {
        int d = i >> 6, t = i & 63;
        sVt[d * 64 + t] = base[(long)t * 1536 + 1024 + h * 64 + d];
    }
    __syncthreads();

    // scores = qn*scale @ kn^T
    v8f acc[4] = {};
    #pragma unroll
    for (int k0 = 0; k0 < 64; k0 += 32) {
        Frag af;
        af.u[0] = *(const u32x4*)&sQ[(16 * wv + nl) * 64 + k0 + hi2 * 8];
        af.u[1] = *(const u32x4*)&sQ[(16 * wv + nl) * 64 + k0 + hi2 * 8 + 16];
        #pragma unroll
        for (int j = 0; j < 4; j++) {
            Frag bf;
            bf.u[0] = *(const u32x4*)&sK[(j * 16 + nl) * 64 + k0 + hi2 * 8];
            bf.u[1] = *(const u32x4*)&sK[(j * 16 + nl) * 64 + k0 + hi2 * 8 + 16];
            acc[j] = wmma_f16(af, bf, acc[j]);
        }
    }
    // register softmax; reductions stay within each 16-lane half (one row each)
    #pragma unroll
    for (int r = 0; r < 8; r++) {
        float mx = -1e30f;
        #pragma unroll
        for (int j = 0; j < 4; j++) mx = fmaxf(mx, acc[j][r]);
        for (int sft = 8; sft >= 1; sft >>= 1) mx = fmaxf(mx, __shfl_xor(mx, sft, 32));
        float sum = 0.f;
        #pragma unroll
        for (int j = 0; j < 4; j++) { float e = __expf(acc[j][r] - mx); acc[j][r] = e; sum += e; }
        for (int sft = 8; sft >= 1; sft >>= 1) sum += __shfl_xor(sum, sft, 32);
        float rinv = 1.f / sum;
        int m = 16 * wv + r + hi2 * 8;
        #pragma unroll
        for (int j = 0; j < 4; j++) sQ[m * 64 + j * 16 + nl] = (f16)(acc[j][r] * rinv);
    }
    __syncthreads();
    // O = P @ V
    v8f o[4] = {};
    #pragma unroll
    for (int k0 = 0; k0 < 64; k0 += 32) {
        Frag af;
        af.u[0] = *(const u32x4*)&sQ[(16 * wv + nl) * 64 + k0 + hi2 * 8];
        af.u[1] = *(const u32x4*)&sQ[(16 * wv + nl) * 64 + k0 + hi2 * 8 + 16];
        #pragma unroll
        for (int j = 0; j < 4; j++) {
            Frag bf;
            bf.u[0] = *(const u32x4*)&sVt[(j * 16 + nl) * 64 + k0 + hi2 * 8];
            bf.u[1] = *(const u32x4*)&sVt[(j * 16 + nl) * 64 + k0 + hi2 * 8 + 16];
            o[j] = wmma_f16(af, bf, o[j]);
        }
    }
    #pragma unroll
    for (int j = 0; j < 4; j++) {
        #pragma unroll
        for (int r = 0; r < 8; r++) {
            int m = 16 * wv + r + hi2 * 8;
            ct[((long)(b * 64 + m)) * 512 + h * 64 + j * 16 + nl] += o[j][r];
        }
    }
}

// ---------- window cosine attention (per (window, head)), N=65 padded to 80/96 ----------
__global__ __launch_bounds__(160)
void win_attn_kernel(const f16* __restrict__ qkv, const float* __restrict__ ls,
                     const float* __restrict__ bias, f16* __restrict__ outp)
{
    __shared__ __align__(16) f16 sQ [80 * 64];
    __shared__ __align__(16) f16 sK [80 * 64];
    __shared__ __align__(16) f16 sVt[64 * 96];
    __shared__ __align__(16) f16 sP [80 * 96];
    const int w = blockIdx.x >> 3, h = blockIdx.x & 7;
    const int tid = threadIdx.x, wv = tid >> 5, lane = tid & 31;
    const int nl = lane & 15, hi2 = lane >> 4;
    const float scale = __expf(fminf(ls[h], 4.6052f));
    const f16* base = qkv + (long)w * NTOK * 1536;

    if (tid < 80) {
        int t = tid;
        if (t < 65) {
            const f16* qr = base + (long)t * 1536 + h * 64;
            float ss = 0.f;
            for (int d = 0; d < 64; d++) { float v = (float)qr[d]; ss += v * v; }
            float inv = scale / fmaxf(sqrtf(ss), 1e-12f);
            for (int d = 0; d < 64; d++) sQ[t * 64 + d] = (f16)((float)qr[d] * inv);
        } else {
            for (int d = 0; d < 64; d++) sQ[t * 64 + d] = (f16)0.f;
        }
    } else {
        int t = tid - 80;
        if (t < 65) {
            const f16* kr = base + (long)t * 1536 + 512 + h * 64;
            float ss = 0.f;
            for (int d = 0; d < 64; d++) { float v = (float)kr[d]; ss += v * v; }
            float inv = 1.f / fmaxf(sqrtf(ss), 1e-12f);
            for (int d = 0; d < 64; d++) sK[t * 64 + d] = (f16)((float)kr[d] * inv);
        } else {
            for (int d = 0; d < 64; d++) sK[t * 64 + d] = (f16)0.f;
        }
    }
    for (int i = tid; i < 64 * 96; i += 160) {
        int d = i / 96, j = i % 96;
        sVt[i] = (j < 65) ? base[(long)j * 1536 + 1024 + h * 64 + d] : (f16)0.f;
    }
    for (int i = tid; i < 80 * 16; i += 160) {        // zero K-pad cols 80..95 of P
        int r = i >> 4, c = i & 15;
        sP[r * 96 + 80 + c] = (f16)0.f;
    }
    __syncthreads();

    v8f acc[5] = {};
    #pragma unroll
    for (int k0 = 0; k0 < 64; k0 += 32) {
        Frag af;
        af.u[0] = *(const u32x4*)&sQ[(16 * wv + nl) * 64 + k0 + hi2 * 8];
        af.u[1] = *(const u32x4*)&sQ[(16 * wv + nl) * 64 + k0 + hi2 * 8 + 16];
        #pragma unroll
        for (int j = 0; j < 5; j++) {
            Frag bf;
            bf.u[0] = *(const u32x4*)&sK[(j * 16 + nl) * 64 + k0 + hi2 * 8];
            bf.u[1] = *(const u32x4*)&sK[(j * 16 + nl) * 64 + k0 + hi2 * 8 + 16];
            acc[j] = wmma_f16(af, bf, acc[j]);
        }
    }
    #pragma unroll
    for (int r = 0; r < 8; r++) {
        int m = 16 * wv + r + hi2 * 8;                // 0..79
        float mx = -1e30f;
        #pragma unroll
        for (int j = 0; j < 5; j++) {
            int n = j * 16 + nl;
            float v = acc[j][r];
            if (n < 65 && m < 65) v += bias[((h * 65) + m) * 65 + n];
            if (n >= 65) v = -1e30f;
            acc[j][r] = v;
            mx = fmaxf(mx, v);
        }
        for (int sft = 8; sft >= 1; sft >>= 1) mx = fmaxf(mx, __shfl_xor(mx, sft, 32));
        float sum = 0.f;
        #pragma unroll
        for (int j = 0; j < 5; j++) { float e = __expf(acc[j][r] - mx); acc[j][r] = e; sum += e; }
        for (int sft = 8; sft >= 1; sft >>= 1) sum += __shfl_xor(sum, sft, 32);
        float rinv = (m < 65) ? 1.f / sum : 0.f;
        #pragma unroll
        for (int j = 0; j < 5; j++) sP[m * 96 + j * 16 + nl] = (f16)(acc[j][r] * rinv);
    }
    __syncthreads();

    v8f o[4] = {};
    #pragma unroll
    for (int k0 = 0; k0 < 96; k0 += 32) {
        Frag af;
        af.u[0] = *(const u32x4*)&sP[(16 * wv + nl) * 96 + k0 + hi2 * 8];
        af.u[1] = *(const u32x4*)&sP[(16 * wv + nl) * 96 + k0 + hi2 * 8 + 16];
        #pragma unroll
        for (int j = 0; j < 4; j++) {
            Frag bf;
            bf.u[0] = *(const u32x4*)&sVt[(j * 16 + nl) * 96 + k0 + hi2 * 8];
            bf.u[1] = *(const u32x4*)&sVt[(j * 16 + nl) * 96 + k0 + hi2 * 8 + 16];
            o[j] = wmma_f16(af, bf, o[j]);
        }
    }
    #pragma unroll
    for (int j = 0; j < 4; j++) {
        #pragma unroll
        for (int r = 0; r < 8; r++) {
            int m = 16 * wv + r + hi2 * 8;
            if (m < 65)
                outp[((long)w * NTOK + m) * 512 + h * 64 + j * 16 + nl] = (f16)o[j][r];
        }
    }
}

// ---------- CPB MLP: tb[225][8] ----------
__global__ void cpb_tb_kernel(const float* __restrict__ w1, const float* __restrict__ b1,
                              const float* __restrict__ w2, float* __restrict__ tb)
{
    int r = blockIdx.x * blockDim.x + threadIdx.x;
    if (r >= 225) return;
    float c0 = (float)(r / 15) - 7.f, c1 = (float)(r % 15) - 7.f;
    float t0v = c0 * (8.f / 7.f), t1v = c1 * (8.f / 7.f);
    float t0 = (t0v < 0.f ? -1.f : (t0v > 0.f ? 1.f : 0.f)) * log2f(fabsf(t0v) + 1.f) * (1.f / 3.f);
    float t1 = (t1v < 0.f ? -1.f : (t1v > 0.f ? 1.f : 0.f)) * log2f(fabsf(t1v) + 1.f) * (1.f / 3.f);
    float acc[8] = {0, 0, 0, 0, 0, 0, 0, 0};
    for (int j = 0; j < 512; j++) {
        float hv = fmaxf(t0 * w1[j] + t1 * w1[512 + j] + b1[j], 0.f);
        #pragma unroll
        for (int hh = 0; hh < 8; hh++) acc[hh] += hv * w2[j * 8 + hh];
    }
    for (int hh = 0; hh < 8; hh++) tb[r * 8 + hh] = acc[hh];
}

// ---------- full bias [8][65][65] with leading zero row/col ----------
__global__ void bias_build_kernel(const float* __restrict__ tb, float* __restrict__ bias)
{
    int idx = blockIdx.x * 256 + threadIdx.x;
    if (idx >= 8 * 65 * 65) return;
    int h = idx / (65 * 65);
    int rem = idx % (65 * 65);
    int i = rem / 65, j = rem % 65;
    float v = 0.f;
    if (i > 0 && j > 0) {
        int ii = i - 1, jj = j - 1;
        int yi = ii >> 3, xi = ii & 7, yj = jj >> 3, xj = jj & 7;
        int rp = (yi - yj + 7) * 15 + (xi - xj + 7);
        float t = tb[rp * 8 + h];
        v = 16.f / (1.f + __expf(-t));
    }
    bias[idx] = v;
}

// ---------- z = concat(ct, window_partition(x)) ----------
__global__ void build_z_kernel(const float* __restrict__ ct, const float* __restrict__ x,
                               float* __restrict__ z)
{
    long i = (long)blockIdx.x * 256 + threadIdx.x;
    if (i >= (long)MWIN * CDIM) return;
    long row = i >> 9; int c = (int)(i & 511);
    int w = (int)(row / NTOK), j = (int)(row % NTOK);
    float v;
    if (j == 0) {
        v = ct[(long)w * 512 + c];
    } else {
        int b = w >> 6, wy = (w >> 3) & 7, wx = w & 7;
        int r = (j - 1) >> 3, cc = (j - 1) & 7;
        v = x[(((long)(b * 64 + wy * 8 + r)) * 64 + (wx * 8 + cc)) * 512 + c];
    }
    z[i] = v;
}

// ---------- window reverse: out = z[:,1:,:] ----------
__global__ void unbuild_kernel(const float* __restrict__ z, float* __restrict__ out)
{
    long i = (long)blockIdx.x * 256 + threadIdx.x;
    if (i >= (long)NB * 64 * 64 * 512) return;
    int c = (int)(i & 511); long t = i >> 9;
    int xc = (int)(t & 63); t >>= 6;
    int y = (int)(t & 63);
    int b = (int)(t >> 6);
    int w = b * 64 + (y >> 3) * 8 + (xc >> 3);
    int j = 1 + (y & 7) * 8 + (xc & 7);
    out[i] = z[((long)w * NTOK + j) * 512 + c];
}

// ---------- launcher ----------
extern "C" void kernel_launch(void* const* d_in, const int* in_sizes, int n_in,
                              void* d_out, int out_size, void* d_ws, size_t ws_size,
                              hipStream_t stream)
{
    (void)in_sizes; (void)n_in; (void)out_size; (void)ws_size;
    const float* x         = (const float*)d_in[0];
    const float* ct        = (const float*)d_in[1];
    const float* w_qkv_ct  = (const float*)d_in[2];
    const float* ls_ct     = (const float*)d_in[3];
    const float* g_ct1     = (const float*)d_in[4];
    const float* b_ct1     = (const float*)d_in[5];
    const float* g_ct2     = (const float*)d_in[6];
    const float* b_ct2     = (const float*)d_in[7];
    const float* mlp_ct_w1 = (const float*)d_in[8];
    const float* mlp_ct_b1 = (const float*)d_in[9];
    const float* mlp_ct_w2 = (const float*)d_in[10];
    const float* mlp_ct_b2 = (const float*)d_in[11];
    const float* w_qkv_win = (const float*)d_in[12];
    const float* w_proj    = (const float*)d_in[13];
    const float* b_proj    = (const float*)d_in[14];
    const float* ls_win    = (const float*)d_in[15];
    const float* g_w1      = (const float*)d_in[16];
    const float* b_w1      = (const float*)d_in[17];
    const float* g_w2      = (const float*)d_in[18];
    const float* b_w2      = (const float*)d_in[19];
    const float* cpb_w1    = (const float*)d_in[20];
    const float* cpb_b1    = (const float*)d_in[21];
    const float* cpb_w2    = (const float*)d_in[22];
    const float* mlp_w_w1  = (const float*)d_in[23];
    const float* mlp_w_b1  = (const float*)d_in[24];
    const float* mlp_w_w2  = (const float*)d_in[25];
    const float* mlp_w_b2  = (const float*)d_in[26];
    float* out = (float*)d_out;

    char* ws = (char*)d_ws;
    size_t off = 0;
    auto alloc = [&](size_t bytes) -> char* {
        char* p = ws + off;
        off += (bytes + 255) & ~(size_t)255;
        return p;
    };
    f16* wt_qkv_ct  = (f16*)alloc((size_t)1536 * 512 * 2);
    f16* wt_mlp_ct1 = (f16*)alloc((size_t)2048 * 512 * 2);
    f16* wt_mlp_ct2 = (f16*)alloc((size_t)512 * 2048 * 2);
    f16* wt_qkv_win = (f16*)alloc((size_t)1536 * 512 * 2);
    f16* wt_proj    = (f16*)alloc((size_t)512 * 512 * 2);
    f16* wt_mlp_w1  = (f16*)alloc((size_t)2048 * 512 * 2);
    f16* wt_mlp_w2  = (f16*)alloc((size_t)512 * 2048 * 2);
    float* ct_cur   = (float*)alloc((size_t)BCT * CDIM * 4);
    f16*  h16       = (f16*)alloc((size_t)MWIN * CDIM * 2);
    f16*  big16     = (f16*)alloc((size_t)MWIN * HIDDEN * 2);   // qkv16 / hid16 aliased
    f16*  qkv16     = big16;
    f16*  hid16     = big16;
    f16*  attn16    = (f16*)alloc((size_t)MWIN * CDIM * 2);
    float* z        = (float*)alloc((size_t)MWIN * CDIM * 4);
    float* cosT     = (float*)alloc(64 * 32 * 4);
    float* sinT     = (float*)alloc(64 * 32 * 4);
    float* tb       = (float*)alloc(225 * 8 * 4);
    float* biasW    = (float*)alloc(8 * 65 * 65 * 4);

    auto cdiv = [](long a, long b) { return (int)((a + b - 1) / b); };

    // weights -> f16 transposed
    wcvt_kernel<<<cdiv(512L * 1536, 256), 256, 0, stream>>>(w_qkv_ct,  wt_qkv_ct,  512, 1536);
    wcvt_kernel<<<cdiv(512L * 2048, 256), 256, 0, stream>>>(mlp_ct_w1, wt_mlp_ct1, 512, 2048);
    wcvt_kernel<<<cdiv(2048L * 512, 256), 256, 0, stream>>>(mlp_ct_w2, wt_mlp_ct2, 2048, 512);
    wcvt_kernel<<<cdiv(512L * 1536, 256), 256, 0, stream>>>(w_qkv_win, wt_qkv_win, 512, 1536);
    wcvt_kernel<<<cdiv(512L * 512, 256), 256, 0, stream>>>(w_proj,    wt_proj,    512, 512);
    wcvt_kernel<<<cdiv(512L * 2048, 256), 256, 0, stream>>>(mlp_w_w1,  wt_mlp_w1,  512, 2048);
    wcvt_kernel<<<cdiv(2048L * 512, 256), 256, 0, stream>>>(mlp_w_w2,  wt_mlp_w2,  2048, 512);

    hipMemcpyAsync(ct_cur, ct, (size_t)BCT * CDIM * 4, hipMemcpyDeviceToDevice, stream);
    rope_tab_kernel<<<cdiv(64 * 32, 256), 256, 0, stream>>>(cosT, sinT);
    cpb_tb_kernel<<<1, 256, 0, stream>>>(cpb_w1, cpb_b1, cpb_w2, tb);
    bias_build_kernel<<<cdiv(8 * 65 * 65, 256), 256, 0, stream>>>(tb, biasW);

    // ---- carrier-token branch ----
    ln_kernel<<<BCT, 128, 0, stream>>>(ct_cur, g_ct1, b_ct1, h16);
    gemm_f16_kernel<GF_OUTF16><<<dim3(BCT / 64, 1536 / 64), 128, 0, stream>>>(
        h16, wt_qkv_ct, nullptr, qkv16, nullptr, BCT, 1536, 512);
    rope_apply_kernel<<<cdiv((long)BCT * 8 * 32 * 2, 256), 256, 0, stream>>>(qkv16, cosT, sinT);
    ct_attn_kernel<<<NB * HEADS, 128, 0, stream>>>(qkv16, ls_ct, ct_cur);
    ln_kernel<<<BCT, 128, 0, stream>>>(ct_cur, g_ct2, b_ct2, h16);
    gemm_f16_kernel<GF_BIAS | GF_GELU | GF_OUTF16><<<dim3(BCT / 64, 2048 / 64), 128, 0, stream>>>(
        h16, wt_mlp_ct1, mlp_ct_b1, hid16, nullptr, BCT, 2048, 512);
    gemm_f16_kernel<GF_BIAS | GF_RESID><<<dim3(BCT / 64, 512 / 64), 128, 0, stream>>>(
        hid16, wt_mlp_ct2, mlp_ct_b2, nullptr, ct_cur, BCT, 512, 2048);

    // ---- window branch ----
    build_z_kernel<<<cdiv((long)MWIN * CDIM, 256), 256, 0, stream>>>(ct_cur, x, z);
    ln_kernel<<<MWIN, 128, 0, stream>>>(z, g_w1, b_w1, h16);
    gemm_f16_kernel<GF_OUTF16><<<dim3(MWIN / 64, 1536 / 64), 128, 0, stream>>>(
        h16, wt_qkv_win, nullptr, qkv16, nullptr, MWIN, 1536, 512);
    win_attn_kernel<<<NWIN * HEADS, 160, 0, stream>>>(qkv16, ls_win, biasW, attn16);
    gemm_f16_kernel<GF_BIAS | GF_RESID><<<dim3(MWIN / 64, 512 / 64), 128, 0, stream>>>(
        attn16, wt_proj, b_proj, nullptr, z, MWIN, 512, 512);
    ln_kernel<<<MWIN, 128, 0, stream>>>(z, g_w2, b_w2, h16);
    gemm_f16_kernel<GF_BIAS | GF_GELU | GF_OUTF16><<<dim3(MWIN / 64, 2048 / 64), 128, 0, stream>>>(
        h16, wt_mlp_w1, mlp_w_b1, hid16, nullptr, MWIN, 2048, 512);
    gemm_f16_kernel<GF_BIAS | GF_RESID><<<dim3(MWIN / 64, 512 / 64), 128, 0, stream>>>(
        hid16, wt_mlp_w2, mlp_w_b2, nullptr, z, MWIN, 512, 2048);

    unbuild_kernel<<<cdiv((long)NB * 64 * 64 * 512, 256), 256, 0, stream>>>(z, out);
}
